// MultiHeadAttention_661424963642
// MI455X (gfx1250) — matrix operand order repaired and checked by reading, not compiled, not run on hardware
//
#include <hip/hip_runtime.h>

// ---------------------------------------------------------------- constants
#define BSZ      2
#define SEQ      2048
#define DIM      2048
#define N_HEADS  32
#define N_KV     8
#define HEAD_DIM 64
#define KV_DIM   (N_KV * HEAD_DIM)   // 512
#define ROWS     (BSZ * SEQ)         // 4096

typedef __attribute__((ext_vector_type(16))) __bf16 v16bf;
typedef __attribute__((ext_vector_type(8)))  float  v8f;

// ---------------------------------------------------------------- helpers
__device__ inline v8f wmma_bf16(v16bf a, v16bf b, v8f c) {
  return __builtin_amdgcn_wmma_f32_16x16x32_bf16(
      /*neg_a=*/false, a, /*neg_b=*/false, b,
      /*c_mod=*/(short)0, c, /*reuse_a=*/false, /*reuse_b=*/false);
}

// A operand (16x32 bf16): lane L holds row M=L%16; elements 0..7 = K kb..kb+7,
// elements 8..15 = K kb+16..kb+23 with kb=(L/16)*8.  Caller passes p = &row[k + kb].
__device__ inline v16bf load_a_tile(const __bf16* p) {
  union { v16bf v; uint4 q[2]; } u;
  u.q[0] = *(const uint4*)(p);
  u.q[1] = *(const uint4*)(p + 16);
  return u.v;
}

// B operand (32x16 bf16): lane L holds column N=L%16; elements 0..7 = K kb..kb+7,
// elements 8..15 = K kb+16..kb+23 with kb=(L/16)*8.  Caller passes p = &col[k + kb].
__device__ inline v16bf load_b_tile(const __bf16* p) {
  union { v16bf v; uint4 q[2]; } u;
  u.q[0] = *(const uint4*)(p);
  u.q[1] = *(const uint4*)(p + 16);
  return u.v;
}

// ---- CDNA5 async Global->LDS (ASYNCcnt-tracked); 16 bytes per lane.
// vdst = LDS byte address (low 32 bits of generic shared pointer), vaddr = 64-bit
// global address.  Drained with s_wait_asynccnt 0.
__device__ inline void async_g2l_b128(unsigned lds_addr, const void* gptr) {
  asm volatile("global_load_async_to_lds_b128 %0, %1, off"
               :
               : "v"(lds_addr), "v"(gptr)
               : "memory");
}
__device__ inline void wait_async0() {
  asm volatile("s_wait_asynccnt 0" ::: "memory");
}
__device__ inline unsigned lds_off(const void* p) {
  return (unsigned)(size_t)p;   // generic LDS address: low 32 bits = LDS offset
}

// ---------------------------------------------------------------- casts
__global__ __launch_bounds__(256) void cast_f32_bf16(const float* __restrict__ in,
                                                     __bf16* __restrict__ out,
                                                     unsigned long long n) {
  unsigned long long i = (unsigned long long)blockIdx.x * blockDim.x + threadIdx.x;
  unsigned long long stride = (unsigned long long)gridDim.x * blockDim.x;
  for (; i < n; i += stride) out[i] = (__bf16)in[i];
}

// ---------------------------------------------------------------- GEMM: C = A * B^T
// A: [M,K] bf16 row-major.  B: [N,K] bf16 row-major (weights, out x in).
// Per wave: 32(M) x 64(N) tile; 8 waves/block; weights are L2-resident (192MB L2).
template <typename OUT_T>
__global__ __launch_bounds__(256) void gemm_xwt(const __bf16* __restrict__ A,
                                                const __bf16* __restrict__ B,
                                                OUT_T* __restrict__ C,
                                                int M, int N, int K) {
  const int lane = threadIdx.x & 31;
  const int wid  = threadIdx.x >> 5;
  const int lm   = lane & 15;
  const int half = lane >> 4;

  const int tile = blockIdx.x * 8 + wid;
  const int ntn  = N >> 6;                 // tiles along N (64 wide)
  const int mt   = tile / ntn;
  const int nt   = tile - mt * ntn;
  const int row0 = mt * 32;
  const int col0 = nt * 64;

  const __bf16* arow0 = A + (size_t)(row0 + lm) * K;
  const __bf16* arow1 = A + (size_t)(row0 + 16 + lm) * K;

  v8f acc[2][4] = {};

  for (int k = 0; k < K; k += 32) {
    __builtin_prefetch(arow0 + k + 256, 0, 0);   // global_prefetch_b8
    __builtin_prefetch(arow1 + k + 256, 0, 0);
    v16bf a0 = load_a_tile(arow0 + k + half * 8);
    v16bf a1 = load_a_tile(arow1 + k + half * 8);
#pragma unroll
    for (int j = 0; j < 4; ++j) {
      const __bf16* bp = B + (size_t)(col0 + j * 16 + lm) * K + k + half * 8;
      v16bf bt = load_b_tile(bp);
      acc[0][j] = wmma_bf16(a0, bt, acc[0][j]);
      acc[1][j] = wmma_bf16(a1, bt, acc[1][j]);
    }
  }

#pragma unroll
  for (int j = 0; j < 4; ++j) {
    const int n = col0 + j * 16 + lm;
#pragma unroll
    for (int r = 0; r < 8; ++r) {
      const int m = row0 + half * 8 + r;
      C[(size_t)m * N + n]        = (OUT_T)acc[0][j][r];
      C[(size_t)(m + 16) * N + n] = (OUT_T)acc[1][j][r];
    }
  }
}

// ---------------------------------------------------------------- RoPE (in place, bf16)
// buf: [ROWS, nheads*64]; thread per (row, head, d<32).
__global__ __launch_bounds__(256) void rope_kernel(__bf16* __restrict__ buf,
                                                   const float* __restrict__ cosb,
                                                   const float* __restrict__ sinb,
                                                   int nheads) {
  unsigned long long i = (unsigned long long)blockIdx.x * blockDim.x + threadIdx.x;
  const int d = (int)(i & 31);
  unsigned long long t = i >> 5;
  const int h = (int)(t % nheads);
  const unsigned long long row = t / nheads;
  const int s = (int)(row & (SEQ - 1));
  const size_t base = row * (size_t)(nheads * HEAD_DIM) + h * HEAD_DIM;

  const float x1 = (float)buf[base + d];
  const float x2 = (float)buf[base + d + 32];
  const float c1 = cosb[s * HEAD_DIM + d],      s1 = sinb[s * HEAD_DIM + d];
  const float c2 = cosb[s * HEAD_DIM + d + 32], s2 = sinb[s * HEAD_DIM + d + 32];
  buf[base + d]      = (__bf16)(x1 * c1 - x2 * s1);
  buf[base + d + 32] = (__bf16)(x2 * c2 + x1 * s2);
}

// ---------------------------------------------------------------- V transpose
// V: [b*SEQ + s][kvh*64 + d]  ->  Vt: [(b*N_KV + kvh)*64 + d][s]
__global__ __launch_bounds__(256) void transpose_v(const __bf16* __restrict__ V,
                                                   __bf16* __restrict__ Vt) {
  unsigned long long i = (unsigned long long)blockIdx.x * blockDim.x + threadIdx.x;
  const int d = (int)(i & 63);
  unsigned long long t = i >> 6;
  const int kvh = (int)(t & 7);
  t >>= 3;
  const int s = (int)(t & (SEQ - 1));
  const int b = (int)(t >> 11);
  const __bf16 val = V[i];
  Vt[((size_t)((b * N_KV + kvh) * HEAD_DIM + d)) * SEQ + s] = val;
}

// ---------------------------------------------------------------- flash attention
// Block = 8 waves = 8 heads sharing one 16-query tile; heads map to 2 KV heads.
// Per 32-key chunk, each KV-head group (4 waves / 128 threads) async-copies its
// K chunk [32x64] and V^T chunk [64x32] into LDS ONCE (4x traffic reduction),
// then all waves run QK^T and PV WMMAs out of LDS.
__global__ __launch_bounds__(256) void flash_attn(const __bf16* __restrict__ Q,
                                                  const __bf16* __restrict__ K,
                                                  const __bf16* __restrict__ Vt,
                                                  __bf16* __restrict__ O) {
  __shared__ alignas(16) __bf16 ldsK[2][32 * 64];   // [group][key][dim]   4KB each
  __shared__ alignas(16) __bf16 ldsV[2][64 * 32];   // [group][dim][key]   4KB each
  __shared__ alignas(16) __bf16 pbuf[8][16 * 32];   // per-wave P staging  1KB each

  const int lane = threadIdx.x & 31;
  const int wid  = threadIdx.x >> 5;
  const int lm   = lane & 15;
  const int half = lane >> 4;

  const int blk = blockIdx.x;           // 0..1023 = b(2) x qt(128) x hg(4)
  const int b   = blk >> 9;
  const int rem = blk & 511;
  const int qt  = rem >> 2;
  const int hg  = rem & 3;
  const int h   = hg * 8 + wid;
  const int g   = wid >> 2;             // KV-head group within block (0/1)
  const int kvh = (h >> 2);             // == hg*2 + g
  const int q0  = qt * 16;

  const int tg   = threadIdx.x & 127;   // thread id within KV-head group
  const int kkey = tg >> 2, kseg = tg & 3;   // K staging: 32B per thread
  const int vdim = tg >> 1, vseg = tg & 1;   // V staging: 32B per thread
  const __bf16* gK0 = K + ((size_t)(b * SEQ + kkey)) * KV_DIM + kvh * HEAD_DIM + kseg * 16;
  const __bf16* gV0 = Vt + ((size_t)((b * N_KV + kvh) * HEAD_DIM + vdim)) * SEQ + vseg * 16;
  const unsigned lK = lds_off(&ldsK[g][kkey * 64 + kseg * 16]);
  const unsigned lV = lds_off(&ldsV[g][vdim * 32 + vseg * 16]);

  // Q A-operand tiles (k = 0..31 and 32..63), row = q0 + lm
  const __bf16* qbase = Q + ((size_t)(b * SEQ + q0 + lm)) * DIM + h * HEAD_DIM;
  const v16bf qa0 = load_a_tile(qbase + half * 8);
  const v16bf qa1 = load_a_tile(qbase + 32 + half * 8);

  float m_run[8], l_run[8];
  v8f o[4] = {};
#pragma unroll
  for (int r = 0; r < 8; ++r) { m_run[r] = -3.0e38f; l_run[r] = 0.0f; }

  const int kmax = q0 + 15;
  const int nch  = (kmax >> 5) + 1;     // 32-key chunks (uniform across block)

  for (int kc = 0; kc < nch; ++kc) {
    const int n0 = kc * 32;

    // ---- stage K / V^T chunks into LDS via async Global->LDS DMA
    __syncthreads();                     // previous iteration's LDS readers done
    async_g2l_b128(lK,      gK0 + (size_t)n0 * KV_DIM);
    async_g2l_b128(lK + 16, gK0 + (size_t)n0 * KV_DIM + 8);
    async_g2l_b128(lV,      gV0 + n0);
    async_g2l_b128(lV + 16, gV0 + n0 + 8);
    wait_async0();                       // each wave drains its own ASYNCcnt
    __syncthreads();                     // staged chunk visible block-wide

    // ---- scores: S = Q * K^T  (two 16x16 C tiles over this 32-key chunk)
    v8f s[2];
#pragma unroll
    for (int j = 0; j < 2; ++j) {
      const __bf16* kb = &ldsK[g][(j * 16 + lm) * 64 + half * 8];
      v16bf b0 = load_b_tile(kb);        // dims 0..31
      v16bf b1 = load_b_tile(kb + 32);   // dims 32..63
      v8f z = {};
      z = wmma_bf16(qa0, b0, z);
      z = wmma_bf16(qa1, b1, z);
      s[j] = z;
    }

    // ---- online softmax update (rows live in lane-16 groups of the C layout)
    float alpha[8];
#pragma unroll
    for (int r = 0; r < 8; ++r) {
      const int qrow = q0 + half * 8 + r;
      const int key0 = n0 + lm;
      const int key1 = n0 + 16 + lm;
      float v0 = s[0][r] * 0.125f; if (key0 > qrow) v0 = -3.0e38f;
      float v1 = s[1][r] * 0.125f; if (key1 > qrow) v1 = -3.0e38f;
      float mx = fmaxf(v0, v1);
      mx = fmaxf(mx, __shfl_xor(mx, 1, 32));
      mx = fmaxf(mx, __shfl_xor(mx, 2, 32));
      mx = fmaxf(mx, __shfl_xor(mx, 4, 32));
      mx = fmaxf(mx, __shfl_xor(mx, 8, 32));
      const float mnew = fmaxf(m_run[r], mx);
      alpha[r] = __expf(m_run[r] - mnew);
      const float p0 = __expf(v0 - mnew);
      const float p1 = __expf(v1 - mnew);
      s[0][r] = p0;
      s[1][r] = p1;
      float ps = p0 + p1;
      ps += __shfl_xor(ps, 1, 32);
      ps += __shfl_xor(ps, 2, 32);
      ps += __shfl_xor(ps, 4, 32);
      ps += __shfl_xor(ps, 8, 32);
      l_run[r] = l_run[r] * alpha[r] + ps;
      m_run[r] = mnew;
    }

    // ---- cross-lane transpose C-layout P -> A-layout via LDS
#pragma unroll
    for (int r = 0; r < 8; ++r) {
      const int row = half * 8 + r;
      pbuf[wid][row * 32 + lm]      = (__bf16)s[0][r];
      pbuf[wid][row * 32 + 16 + lm] = (__bf16)s[1][r];
    }
    __syncthreads();                     // pbuf visible (ds drained by barrier)
    const v16bf pa = load_a_tile(&pbuf[wid][lm * 32 + half * 8]);

    // ---- O = alpha*O + P * V  (four 16-dim output tiles, V from LDS)
#pragma unroll
    for (int jd = 0; jd < 4; ++jd) {
      const __bf16* vb = &ldsV[g][(jd * 16 + lm) * 32 + half * 8];
      v16bf bv = load_b_tile(vb);
      v8f oc;
#pragma unroll
      for (int r = 0; r < 8; ++r) oc[r] = o[jd][r] * alpha[r];
      o[jd] = wmma_bf16(pa, bv, oc);
    }
  }

  // ---- normalize and store (row = b*S + q, col = h*64 + d)
#pragma unroll
  for (int jd = 0; jd < 4; ++jd) {
#pragma unroll
    for (int r = 0; r < 8; ++r) {
      const int qrow = q0 + half * 8 + r;
      const int n = jd * 16 + lm;
      O[((size_t)(b * SEQ + qrow)) * DIM + h * HEAD_DIM + n] =
          (__bf16)(o[jd][r] / l_run[r]);
    }
  }
}

// ---------------------------------------------------------------- launch
extern "C" void kernel_launch(void* const* d_in, const int* in_sizes, int n_in,
                              void* d_out, int out_size, void* d_ws, size_t ws_size,
                              hipStream_t stream) {
  (void)in_sizes; (void)n_in; (void)out_size; (void)ws_size;

  const float* hs   = (const float*)d_in[0];
  /* d_in[1] = additive causal mask: applied analytically */
  const float* cosb = (const float*)d_in[2];
  const float* sinb = (const float*)d_in[3];
  const float* Wq   = (const float*)d_in[4];
  const float* Wk   = (const float*)d_in[5];
  const float* Wv   = (const float*)d_in[6];
  const float* Wo   = (const float*)d_in[7];
  float* out = (float*)d_out;

  // workspace layout (bf16 elements)
  __bf16* p = (__bf16*)d_ws;
  __bf16* Xb  = p; p += (size_t)ROWS * DIM;        // 8.4M
  __bf16* Wqb = p; p += (size_t)DIM * DIM;         // 4.2M
  __bf16* Wkb = p; p += (size_t)KV_DIM * DIM;      // 1.0M
  __bf16* Wvb = p; p += (size_t)KV_DIM * DIM;      // 1.0M
  __bf16* Wob = p; p += (size_t)DIM * DIM;         // 4.2M
  __bf16* Qb  = p; p += (size_t)ROWS * DIM;        // 8.4M
  __bf16* Kb  = p; p += (size_t)ROWS * KV_DIM;     // 2.1M
  __bf16* Vb  = p; p += (size_t)ROWS * KV_DIM;     // 2.1M
  __bf16* Vtb = p; p += (size_t)ROWS * KV_DIM;     // 2.1M
  __bf16* Ab  = p;                                 // 8.4M   (~84 MB total)

  // 1) casts to bf16
  cast_f32_bf16<<<4096, 256, 0, stream>>>(hs, Xb,  (unsigned long long)ROWS * DIM);
  cast_f32_bf16<<<4096, 256, 0, stream>>>(Wq, Wqb, (unsigned long long)DIM * DIM);
  cast_f32_bf16<<<2048, 256, 0, stream>>>(Wk, Wkb, (unsigned long long)KV_DIM * DIM);
  cast_f32_bf16<<<2048, 256, 0, stream>>>(Wv, Wvb, (unsigned long long)KV_DIM * DIM);
  cast_f32_bf16<<<4096, 256, 0, stream>>>(Wo, Wob, (unsigned long long)DIM * DIM);

  // 2) projections  (tiles = (M/32)*(N/64), 8 waves/block)
  gemm_xwt<__bf16><<<(ROWS / 32) * (DIM / 64) / 8,    256, 0, stream>>>(Xb, Wqb, Qb, ROWS, DIM,    DIM);
  gemm_xwt<__bf16><<<(ROWS / 32) * (KV_DIM / 64) / 8, 256, 0, stream>>>(Xb, Wkb, Kb, ROWS, KV_DIM, DIM);
  gemm_xwt<__bf16><<<(ROWS / 32) * (KV_DIM / 64) / 8, 256, 0, stream>>>(Xb, Wvb, Vb, ROWS, KV_DIM, DIM);

  // 3) RoPE
  rope_kernel<<<(ROWS * N_HEADS * 32) / 256, 256, 0, stream>>>(Qb, cosb, sinb, N_HEADS);
  rope_kernel<<<(ROWS * N_KV   * 32) / 256, 256, 0, stream>>>(Kb, cosb, sinb, N_KV);

  // 4) V -> V^T
  transpose_v<<<((size_t)ROWS * KV_DIM) / 256, 256, 0, stream>>>(Vb, Vtb);

  // 5) flash attention: 2(b) * 128(qtile) * 4(head-group) blocks, 8 waves each
  flash_attn<<<1024, 256, 0, stream>>>(Qb, Kb, Vtb, Ab);

  // 6) output projection -> fp32
  gemm_xwt<float><<<(ROWS / 32) * (DIM / 64) / 8, 256, 0, stream>>>(Ab, Wob, out, ROWS, DIM, DIM);
}
